// MultiHeadAttention_19344532701214
// MI455X (gfx1250) — matrix operand order repaired and checked by reading, not compiled, not run on hardware
//
#include <hip/hip_runtime.h>

typedef __attribute__((ext_vector_type(16))) _Float16 v16h;
typedef __attribute__((ext_vector_type(8)))  _Float16 v8h;
typedef __attribute__((ext_vector_type(8)))  float    v8f;
typedef __attribute__((ext_vector_type(4)))  float    v4f;

constexpr int kB  = 2;
constexpr int kS  = 384;
constexpr int kD  = 384;
constexpr int kH  = 12;
constexpr int kDK = 32;
constexpr int kBS = kB * kS;                    // 768 rows
constexpr float kInvSqrtDk = 0.17677669529663687f;  // 1/sqrt(32)

// ---------------------------------------------------------------------------
// WMMA fragment loaders (16-bit; A/B use this code's contiguous k order, not the ISA table order)
// A 16x32: lane&15 = row M; lanes 0-15 hold K 0..15, lanes 16-31 K 16..31.
// B 32x16: lane&15 = col N; lanes 0-15 hold K 0..15, lanes 16-31 K 16..31.
// C 16x16 f32: element (m,n) in lane (n + (m>=8)*16), vgpr m&7.
// ---------------------------------------------------------------------------
__device__ __forceinline__ v16h frag_a16(const _Float16* p) {
  // caller has already applied: row*(ld) + ((lane>>4)<<4)
  return *(const v16h*)(p);
}
__device__ __forceinline__ v16h frag_b16(const _Float16* p) {
  // caller has already applied: col*(ld) + ((lane>>4)<<4)
  return *(const v16h*)(p);
}

__device__ __forceinline__ v8f wmma_f16(v16h a, v16h b, v8f c) {
  return __builtin_amdgcn_wmma_f32_16x16x32_f16(false, a, false, b,
                                                (short)0, c, false, false);
}

// ---------------------------------------------------------------------------
// f32 -> f16 conversion (inputs read once -> non-temporal)
// ---------------------------------------------------------------------------
__global__ void f32_to_f16_kernel(const float* __restrict__ src,
                                  _Float16* __restrict__ dst, int n) {
  int i = blockIdx.x * blockDim.x + threadIdx.x;
  if (i < n) dst[i] = (_Float16)__builtin_nontemporal_load(src + i);
}

// ---------------------------------------------------------------------------
// C = X @ W^T (+bias). X: f16 [M,K] row-major. W: f16 [N,K] row-major.
// mode 0: f16 out [M,N] row-major
// mode 1: f16 out written transposed per-head: Vt[((b*H+h)*DK+dk)*S + s]
// mode 2: f32 out [M,N] (+bias) -> final output
// one wave per 16x16 tile, K-loop of 32.
// ---------------------------------------------------------------------------
__global__ __launch_bounds__(256) void gemm_xwt_kernel(
    const _Float16* __restrict__ X, const _Float16* __restrict__ W,
    const float* __restrict__ bias, void* __restrict__ out,
    int M, int N, int K, int mode) {
  const int lane = threadIdx.x & 31;
  const int wid  = blockIdx.x * (blockDim.x >> 5) + (threadIdx.x >> 5);
  const int numNt = N >> 4;
  const int tiles = (M >> 4) * numNt;
  if (wid >= tiles) return;
  const int m0 = (wid / numNt) << 4;
  const int n0 = (wid % numNt) << 4;

  const _Float16* pa = X + (size_t)(m0 + (lane & 15)) * K + ((lane >> 4) << 4);
  const _Float16* pb = W + (size_t)(n0 + (lane & 15)) * K + ((lane >> 4) << 4);

  v8f c = {};
  for (int kc = 0; kc < K; kc += 32) {
    v16h a = frag_a16(pa + kc);
    v16h b = frag_b16(pb + kc);
    c = wmma_f16(a, b, c);
  }

  const int n  = lane & 15;
  const int mb = (lane >> 4) << 3;
  if (mode == 0) {
    _Float16* o16 = (_Float16*)out;
    float bb = bias ? bias[n0 + n] : 0.0f;
#pragma unroll
    for (int r = 0; r < 8; ++r)
      o16[(size_t)(m0 + mb + r) * N + n0 + n] = (_Float16)(c[r] + bb);
  } else if (mode == 1) {
    _Float16* o16 = (_Float16*)out;
    const int j = n0 + n, hh = j >> 5, dk = j & 31;
    float bb = bias ? bias[j] : 0.0f;
#pragma unroll
    for (int r = 0; r < 8; ++r) {
      int m = m0 + mb + r;
      int bi = m / kS, s = m % kS;
      o16[((size_t)(bi * kH + hh) * kDK + dk) * kS + s] = (_Float16)(c[r] + bb);
    }
  } else {
    float* of = (float*)out;
    float bb = bias ? bias[n0 + n] : 0.0f;
#pragma unroll
    for (int r = 0; r < 8; ++r)
      of[(size_t)(m0 + mb + r) * N + n0 + n] = c[r] + bb;
  }
}

// ---------------------------------------------------------------------------
// Attention core. One block (256 thr = 8 waves) per (b, h, q-tile of 16).
// Phase 1a: scores = Q.K^T via WMMA (24 k-tiles, 1 wmma each).
// Phase 1b: + q . bias_k[b,q,k,h-chunk]  (NT streaming f32, b128 per lane).
// Phase 2 : row softmax (with query-row mask, matching reference broadcast).
// Phase 3a: attn @ V via WMMA (Vt pre-transposed -> contiguous B frags).
// Phase 3b: + sum_k attn * bias_v  (lane = d, NT coalesced stream).
// ---------------------------------------------------------------------------
__global__ __launch_bounds__(256) void attention_kernel(
    const _Float16* __restrict__ Qh, const _Float16* __restrict__ Kh,
    const _Float16* __restrict__ Vt, const float* __restrict__ bias_k,
    const float* __restrict__ bias_v, const int* __restrict__ mask,
    _Float16* __restrict__ Ch) {
  const int numQt = kS / 16;  // 24
  const int qt = blockIdx.x % numQt;
  const int h  = (blockIdx.x / numQt) % kH;
  const int bi = blockIdx.x / (numQt * kH);
  const int q0 = qt << 4;

  const int tid  = threadIdx.x;
  const int lane = tid & 31;
  const int wid  = tid >> 5;

  __shared__ float    sQ[16][kDK];       // q tile, f32 (2 KB)
  __shared__ float    sScore[16][kS];    // scores/attn f32 (24 KB)
  __shared__ _Float16 sAttn[16][kS];     // attn f16 for WMMA A (12 KB)
  __shared__ float    sPart[8][16][16];  // attn@V partials (8 KB)

  // stage Q tile as f32 for the bias dot products
  for (int i = tid; i < 16 * kDK; i += 256) {
    int q = i >> 5, d = i & 31;
    sQ[q][d] = (float)Qh[(size_t)(bi * kS + q0 + q) * kD + h * kDK + d];
  }

  // ---- Phase 1a: Q.K^T ----------------------------------------------------
  {
    const _Float16* pa =
        Qh + (size_t)(bi * kS + q0 + (lane & 15)) * kD + h * kDK + ((lane >> 4) << 4);
    v16h a = frag_a16(pa);
    for (int kt = wid; kt < numQt; kt += 8) {
      const _Float16* pb = Kh + (size_t)(bi * kS + kt * 16 + (lane & 15)) * kD +
                           h * kDK + ((lane >> 4) << 4);
      v8f c = {};
      c = wmma_f16(a, frag_b16(pb), c);
      const int n = lane & 15, mb = (lane >> 4) << 3;
#pragma unroll
      for (int r = 0; r < 8; ++r) sScore[mb + r][kt * 16 + n] = c[r];
    }
  }
  __syncthreads();

  // ---- Phase 1b: + q . bias_k (non-temporal stream), then scale -----------
  for (int p = tid; p < 16 * kS; p += 256) {
    const int q = p / kS, k = p % kS;
    const v4f* bp = (const v4f*)(bias_k +
        ((size_t)(bi * kS + q0 + q) * kS + k) * kD + h * kDK);
    float acc = 0.0f;
#pragma unroll
    for (int d4 = 0; d4 < 8; ++d4) {
      v4f t = __builtin_nontemporal_load(bp + d4);
      acc += t.x * sQ[q][4 * d4 + 0] + t.y * sQ[q][4 * d4 + 1] +
             t.z * sQ[q][4 * d4 + 2] + t.w * sQ[q][4 * d4 + 3];
    }
    sScore[q][k] = (sScore[q][k] + acc) * kInvSqrtDk;
  }
  __syncthreads();

  // ---- Phase 2: softmax over k (query-row mask per reference) -------------
  for (int q = wid; q < 16; q += 8) {
    const int mv = mask[bi * kS + q0 + q];
    float mx = -3.0e38f;
    for (int k = lane; k < kS; k += 32) {
      float s = (mv == 1) ? -1.0e9f : sScore[q][k];
      sScore[q][k] = s;
      mx = fmaxf(mx, s);
    }
#pragma unroll
    for (int off = 16; off > 0; off >>= 1) mx = fmaxf(mx, __shfl_xor(mx, off));
    float sum = 0.0f;
    for (int k = lane; k < kS; k += 32) {
      float e = __expf(sScore[q][k] - mx);
      sScore[q][k] = e;
      sum += e;
    }
#pragma unroll
    for (int off = 16; off > 0; off >>= 1) sum += __shfl_xor(sum, off);
    const float inv = 1.0f / sum;
    for (int k = lane; k < kS; k += 32) {
      float a = sScore[q][k] * inv;
      sScore[q][k] = a;
      sAttn[q][k] = (_Float16)a;
    }
  }
  __syncthreads();

  // ---- Phase 3a: attn @ V (wave w: ntile = w&1, 3 k-chunks each) ----------
  {
    const int ntile = wid & 1;
    const int kstart = (wid >> 1) * 3;  // 12 chunks of K=32 over 4 wave-pairs
    v8f c = {};
    for (int kc = kstart; kc < kstart + 3; ++kc) {
      const _Float16* pa = &sAttn[lane & 15][kc * 32 + ((lane >> 4) << 4)];
      const _Float16* pb = Vt +
          (size_t)((bi * kH + h) * kDK + ntile * 16 + (lane & 15)) * kS +
          kc * 32 + ((lane >> 4) << 4);
      c = wmma_f16(frag_a16(pa), frag_b16(pb), c);
    }
    const int n = lane & 15, mb = (lane >> 4) << 3;
#pragma unroll
    for (int r = 0; r < 8; ++r) sPart[wid][mb + r][n] = c[r];
  }
  __syncthreads();

  // ---- Phase 3b: + sum_k attn * bias_v (lane = d, NT coalesced) -----------
  for (int q = wid; q < 16; q += 8) {
    const float* bp = bias_v + (size_t)(bi * kS + q0 + q) * kS * kD + h * kDK + lane;
    float acc0 = 0.0f, acc1 = 0.0f;
    for (int k = 0; k < kS; k += 8) {
#pragma unroll
      for (int u = 0; u < 8; u += 2) {
        acc0 += sScore[q][k + u + 0] *
                __builtin_nontemporal_load(bp + (size_t)(k + u + 0) * kD);
        acc1 += sScore[q][k + u + 1] *
                __builtin_nontemporal_load(bp + (size_t)(k + u + 1) * kD);
      }
    }
    const float acc = acc0 + acc1;
    const int d = lane, nt = d >> 4, ns = d & 15;
    float r = acc + sPart[nt + 0][q][ns] + sPart[nt + 2][q][ns] +
              sPart[nt + 4][q][ns] + sPart[nt + 6][q][ns];
    Ch[(size_t)(bi * kS + q0 + q) * kD + h * kDK + d] = (_Float16)r;
  }
}

// ---------------------------------------------------------------------------
extern "C" void kernel_launch(void* const* d_in, const int* in_sizes, int n_in,
                              void* d_out, int out_size, void* d_ws, size_t ws_size,
                              hipStream_t stream) {
  const float* q_in   = (const float*)d_in[0];
  const float* k_in   = (const float*)d_in[1];
  const float* v_in   = (const float*)d_in[2];
  const float* bias_k = (const float*)d_in[3];
  const float* bias_v = (const float*)d_in[4];
  const float* Wq = (const float*)d_in[5];
  const float* bq = (const float*)d_in[6];
  const float* Wk = (const float*)d_in[7];
  const float* bk = (const float*)d_in[8];
  const float* Wv = (const float*)d_in[9];
  const float* bv = (const float*)d_in[10];
  const float* Wo = (const float*)d_in[11];
  const float* bo = (const float*)d_in[12];
  const int* mask = (const int*)d_in[13];

  const size_t nAct = (size_t)kBS * kD;   // 294912
  const size_t nWt  = (size_t)kD * kD;    // 147456

  _Float16* ws = (_Float16*)d_ws;
  size_t o = 0;
  _Float16* XQ  = ws + o; o += nAct;
  _Float16* XK  = ws + o; o += nAct;
  _Float16* XV  = ws + o; o += nAct;
  _Float16* Qh  = ws + o; o += nAct;
  _Float16* Kh  = ws + o; o += nAct;
  _Float16* Vt  = ws + o; o += nAct;   // transposed layout [b,h,dk,s]
  _Float16* Ch  = ws + o; o += nAct;
  _Float16* Wqh = ws + o; o += nWt;
  _Float16* Wkh = ws + o; o += nWt;
  _Float16* Wvh = ws + o; o += nWt;
  _Float16* Woh = ws + o; o += nWt;

  auto cvt = [&](const float* s, _Float16* d, int n) {
    f32_to_f16_kernel<<<(n + 255) / 256, 256, 0, stream>>>(s, d, n);
  };
  cvt(q_in, XQ, (int)nAct);
  cvt(k_in, XK, (int)nAct);
  cvt(v_in, XV, (int)nAct);
  cvt(Wq, Wqh, (int)nWt);
  cvt(Wk, Wkh, (int)nWt);
  cvt(Wv, Wvh, (int)nWt);
  cvt(Wo, Woh, (int)nWt);

  const int tiles = (kBS / 16) * (kD / 16);        // 1152 tiles
  const int gemmBlocks = (tiles + 7) / 8;          // 8 waves (16x16 tiles) / block
  gemm_xwt_kernel<<<gemmBlocks, 256, 0, stream>>>(XQ, Wqh, bq, Qh, kBS, kD, kD, 0);
  gemm_xwt_kernel<<<gemmBlocks, 256, 0, stream>>>(XK, Wkh, bk, Kh, kBS, kD, kD, 0);
  gemm_xwt_kernel<<<gemmBlocks, 256, 0, stream>>>(XV, Wvh, bv, Vt, kBS, kD, kD, 1);

  const int attnBlocks = kB * kH * (kS / 16);      // 576
  attention_kernel<<<attnBlocks, 256, 0, stream>>>(Qh, Kh, Vt, bias_k, bias_v,
                                                   mask, Ch);

  gemm_xwt_kernel<<<gemmBlocks, 256, 0, stream>>>(Ch, Woh, bo, (float*)d_out,
                                                  kBS, kD, kD, 2);
}
